// LSTM_15805479649315
// MI455X (gfx1250) — compile-verified, matrix operand order checked by reading, not run on hardware
//
#include <hip/hip_runtime.h>
#include <hip/hip_bf16.h>

typedef __attribute__((ext_vector_type(16))) __bf16 v16bf;
typedef __attribute__((ext_vector_type(8)))  float  v8f;

#define B_ 64
#define S_ 512
#define E_ 256
#define H_ 512
#define O_ 1000
#define K_ 768      // H + E
#define NG 2048     // 4*H packed: n = (h/16)*64 + g*16 + (h%16)
#define KT 24       // K_ / 32
#define LPITCH 40   // LDS row pitch in bf16 (80 bytes = 20 dwords, bank-conflict free)

__device__ __forceinline__ float sigf(float v) {
    return 1.0f / (1.0f + __expf(-v));
}

// generic -> AS(3) -> raw LDS byte offset (what async-to-LDS VDST expects)
typedef __attribute__((address_space(3))) const __bf16 lds_cbf16_t;
__device__ __forceinline__ unsigned lds_off(const __bf16* p) {
    return (unsigned)(unsigned long long)(lds_cbf16_t*)p;
}

// ---------------------------------------------------------------------------
// init: comb0 = [ bf16(hidden) | bf16(embed[x[:,0]]) ], c = c_in, h = hidden
// ---------------------------------------------------------------------------
__global__ void lstm_init(const float* __restrict__ hidden_in,
                          const float* __restrict__ c_in,
                          const float* __restrict__ embed_w,
                          const int*   __restrict__ x,
                          __bf16* __restrict__ comb,
                          float*  __restrict__ c,
                          float*  __restrict__ h)
{
    int idx = blockIdx.x * blockDim.x + threadIdx.x;
    if (idx < B_ * H_) {
        int b = idx / H_, hh = idx % H_;
        float v = hidden_in[idx];
        comb[b * K_ + hh] = (__bf16)v;
        h[idx] = v;
        c[idx] = c_in[idx];
    } else {
        int j = idx - B_ * H_;
        if (j < B_ * E_) {
            int b = j / E_, e = j % E_;
            int tok = x[b * S_ + 0];
            comb[b * K_ + H_ + e] = (__bf16)embed_w[(size_t)tok * E_ + e];
        }
    }
}

// ---------------------------------------------------------------------------
// pack (gate-interleaved, N-major):
//   n = (h/16)*64 + g*16 + (h%16);  Wt[n,k] = bf16(W_g[k,h]);  bcat[n] = b_g[h]
// ---------------------------------------------------------------------------
__global__ void lstm_pack(const float* __restrict__ Wf, const float* __restrict__ Wi,
                          const float* __restrict__ Wc, const float* __restrict__ Wo,
                          const float* __restrict__ bf_, const float* __restrict__ bi_,
                          const float* __restrict__ bc_, const float* __restrict__ bo_,
                          __bf16* __restrict__ Wt, float* __restrict__ bcat)
{
    int idx = blockIdx.x * blockDim.x + threadIdx.x;
    if (idx >= NG * K_) return;
    int n = idx / K_, k = idx % K_;
    int g  = (n >> 4) & 3;
    int hh = (n >> 6) * 16 + (n & 15);
    const float* Wg = (g == 0) ? Wf : (g == 1) ? Wi : (g == 2) ? Wc : Wo;
    Wt[idx] = (__bf16)Wg[(size_t)k * H_ + hh];
    if (k == 0) {
        const float* bg = (g == 0) ? bf_ : (g == 1) ? bi_ : (g == 2) ? bc_ : bo_;
        bcat[n] = bg[hh];
    }
}

// ---------------------------------------------------------------------------
// Fused LSTM step: async-to-LDS double-buffered GEMM (bf16 WMMA, bias-init
// f32 accum) + gates + cell update + bf16(h)->comb_next + t+1 emb gather.
// Block = 128 threads (4 waves); wave m -> rows m*16..m*16+15;
// blockIdx.x = h-block (16 h values); accumulator a = gate a.
// ---------------------------------------------------------------------------
__global__ __launch_bounds__(128)
void lstm_step(const __bf16* __restrict__ comb_cur,
               __bf16* __restrict__ comb_next,
               const __bf16* __restrict__ Wt,
               const float* __restrict__ bcat,
               float* __restrict__ c,
               float* __restrict__ h,
               const float* __restrict__ embed_w,
               const int*   __restrict__ x,
               int t)
{
    __shared__ __align__(16) __bf16 ldsA[2][64][LPITCH];   // comb rows, k-tile
    __shared__ __align__(16) __bf16 ldsB[2][64][LPITCH];   // Wt cols,  k-tile

    const int tid   = threadIdx.x;
    const int lane  = tid & 31;
    const int wave  = tid >> 5;
    const int m0    = wave * 16;
    const int n0    = blockIdx.x * 64;
    const int lrow  = lane & 15;
    const int khalf = (lane >> 4) << 3;   // lanes 16-31 start at K+8

    // accumulators initialized to the (broadcast-over-rows) gate biases
    float b0 = bcat[n0 +  0 + lrow];
    float b1 = bcat[n0 + 16 + lrow];
    float b2 = bcat[n0 + 32 + lrow];
    float b3 = bcat[n0 + 48 + lrow];
    v8f acc0 = {b0, b0, b0, b0, b0, b0, b0, b0};
    v8f acc1 = {b1, b1, b1, b1, b1, b1, b1, b1};
    v8f acc2 = {b2, b2, b2, b2, b2, b2, b2, b2};
    v8f acc3 = {b3, b3, b3, b3, b3, b3, b3, b3};

    // async copy of one 8KB k-tile (A: 64x32, B: 64x32 bf16) = 512 x 16B
    // chunks; thread handles chunks tid+{0,128,256,384}: first two -> A,
    // last two -> B. row = (ch>>2)&63, 16B-quarter q = ch&3.
    auto issue = [&](int buf, int kt) {
        const int kbase = kt * 32;
#pragma unroll
        for (int j = 0; j < 4; ++j) {
            const int ch  = tid + j * 128;
            const int row = (ch >> 2) & 63;
            const int q   = ch & 3;
            if (j < 2) {
                const __bf16* g = comb_cur + (size_t)row * K_ + kbase + q * 8;
                unsigned l = lds_off(&ldsA[buf][row][q * 8]);
                asm volatile("global_load_async_to_lds_b128 %0, %1, off"
                             :: "v"(l), "v"(g) : "memory");
            } else {
                const __bf16* g = Wt + (size_t)(n0 + row) * K_ + kbase + q * 8;
                unsigned l = lds_off(&ldsB[buf][row][q * 8]);
                asm volatile("global_load_async_to_lds_b128 %0, %1, off"
                             :: "v"(l), "v"(g) : "memory");
            }
        }
    };

    union Frag { v16bf v; uint4 u[2]; };

    issue(0, 0);
    asm volatile("s_wait_asynccnt 0x0" ::: "memory");
    __syncthreads();

#pragma unroll
    for (int kt = 0; kt < KT; ++kt) {
        const int buf = kt & 1;
        if (kt + 1 < KT) issue(buf ^ 1, kt + 1);   // overlap fetch with WMMA

        Frag A, B0, B1, B2, B3;
        A.u[0]  = *(const uint4*)&ldsA[buf][m0 + lrow][khalf];
        A.u[1]  = *(const uint4*)&ldsA[buf][m0 + lrow][khalf + 16];
        B0.u[0] = *(const uint4*)&ldsB[buf][ 0 + lrow][khalf];
        B0.u[1] = *(const uint4*)&ldsB[buf][ 0 + lrow][khalf + 16];
        B1.u[0] = *(const uint4*)&ldsB[buf][16 + lrow][khalf];
        B1.u[1] = *(const uint4*)&ldsB[buf][16 + lrow][khalf + 16];
        B2.u[0] = *(const uint4*)&ldsB[buf][32 + lrow][khalf];
        B2.u[1] = *(const uint4*)&ldsB[buf][32 + lrow][khalf + 16];
        B3.u[0] = *(const uint4*)&ldsB[buf][48 + lrow][khalf];
        B3.u[1] = *(const uint4*)&ldsB[buf][48 + lrow][khalf + 16];

        acc0 = __builtin_amdgcn_wmma_f32_16x16x32_bf16(false, A.v, false, B0.v,
                                                       (short)0, acc0, false, false);
        acc1 = __builtin_amdgcn_wmma_f32_16x16x32_bf16(false, A.v, false, B1.v,
                                                       (short)0, acc1, false, false);
        acc2 = __builtin_amdgcn_wmma_f32_16x16x32_bf16(false, A.v, false, B2.v,
                                                       (short)0, acc2, false, false);
        acc3 = __builtin_amdgcn_wmma_f32_16x16x32_bf16(false, A.v, false, B3.v,
                                                       (short)0, acc3, false, false);

        if (kt + 1 < KT) {
            asm volatile("s_wait_asynccnt 0x0" ::: "memory");
            __syncthreads();
        }
    }

    // D layout: element r -> row m0 + r + 8*(lane>=16); h = blockIdx*16 + (lane&15)
    const int hidx  = blockIdx.x * 16 + lrow;
    const int rbase = m0 + ((lane >> 4) << 3);
#pragma unroll
    for (int r = 0; r < 8; ++r) {
        const int row = rbase + r;
        float f = sigf(acc0[r]);
        float i = sigf(acc1[r]);
        float g = tanhf(acc2[r]);
        float o = sigf(acc3[r]);
        const size_t ci = (size_t)row * H_ + hidx;
        float cn = f * c[ci] + i * g;
        c[ci] = cn;
        float hn = o * tanhf(cn);
        comb_next[(size_t)row * K_ + hidx] = (__bf16)hn;
        if (t == S_ - 1) h[ci] = hn;
    }

    // gather embedding for step t+1 into comb_next (4096 threads x 4 elems)
    if (t + 1 < S_) {
        const int gtid = blockIdx.x * 128 + tid;
#pragma unroll
        for (int q = 0; q < 4; ++q) {
            int idx = gtid * 4 + q;              // [0, 16384) == 64*256
            int b = idx >> 8, e = idx & 255;
            int tok = x[b * S_ + t + 1];
            comb_next[(size_t)b * K_ + H_ + e] = (__bf16)embed_w[(size_t)tok * E_ + e];
        }
    }
}

// ---------------------------------------------------------------------------
// dense: out[b,o] = h[b,:] . W_d[:,o] + b_d[o]; also emit hidden (tuple out #2)
// ---------------------------------------------------------------------------
__global__ void lstm_dense(const float* __restrict__ h,
                           const float* __restrict__ Wd,
                           const float* __restrict__ bd,
                           float* __restrict__ out)
{
    int idx = blockIdx.x * blockDim.x + threadIdx.x;
    if (idx < B_ * O_) {
        int b = idx / O_, o = idx % O_;
        const float* hr = h + (size_t)b * H_;
        float s = bd[o];
#pragma unroll 8
        for (int k = 0; k < H_; ++k)
            s += hr[k] * Wd[(size_t)k * O_ + o];
        out[idx] = s;
    } else {
        int j = idx - B_ * O_;
        if (j < B_ * H_)
            out[B_ * O_ + j] = h[j];
    }
}

// ---------------------------------------------------------------------------
extern "C" void kernel_launch(void* const* d_in, const int* in_sizes, int n_in,
                              void* d_out, int out_size, void* d_ws, size_t ws_size,
                              hipStream_t stream)
{
    (void)in_sizes; (void)n_in; (void)out_size; (void)ws_size;

    const int*   x       = (const int*)d_in[0];
    const float* hid_in  = (const float*)d_in[1];
    const float* c_in    = (const float*)d_in[2];
    const float* embed_w = (const float*)d_in[3];
    const float* Wf = (const float*)d_in[4];  const float* bf_ = (const float*)d_in[5];
    const float* Wi = (const float*)d_in[6];  const float* bi_ = (const float*)d_in[7];
    const float* Wc = (const float*)d_in[8];  const float* bc_ = (const float*)d_in[9];
    const float* Wo = (const float*)d_in[10]; const float* bo_ = (const float*)d_in[11];
    const float* Wd = (const float*)d_in[12]; const float* bd_ = (const float*)d_in[13];
    float* out = (float*)d_out;

    char*  ws  = (char*)d_ws;
    size_t off = 0;
    auto alloc = [&](size_t bytes) -> void* {
        void* p = ws + off;
        off = (off + bytes + 255) & ~(size_t)255;
        return p;
    };
    __bf16* combA = (__bf16*)alloc((size_t)B_ * K_ * 2);
    __bf16* combB = (__bf16*)alloc((size_t)B_ * K_ * 2);
    __bf16* Wt    = (__bf16*)alloc((size_t)NG * K_ * 2);
    float*  bcat  = (float*) alloc((size_t)NG * 4);
    float*  cst   = (float*) alloc((size_t)B_ * H_ * 4);
    float*  hst   = (float*) alloc((size_t)B_ * H_ * 4);

    hipLaunchKernelGGL(lstm_init, dim3(192), dim3(256), 0, stream,
                       hid_in, c_in, embed_w, x, combA, cst, hst);
    hipLaunchKernelGGL(lstm_pack, dim3((NG * K_ + 255) / 256), dim3(256), 0, stream,
                       Wf, Wi, Wc, Wo, bf_, bi_, bc_, bo_, Wt, bcat);

    for (int t = 0; t < S_; ++t) {
        __bf16* cur = (t & 1) ? combB : combA;
        __bf16* nxt = (t & 1) ? combA : combB;
        hipLaunchKernelGGL(lstm_step, dim3(H_ / 16), dim3(128), 0, stream,
                           cur, nxt, Wt, bcat, cst, hst, embed_w, x, t);
    }

    hipLaunchKernelGGL(lstm_dense, dim3((B_ * O_ + B_ * H_ + 255) / 256), dim3(256),
                       0, stream, hst, Wd, bd_, out);
}
